// AFRM_1245540516473
// MI455X (gfx1250) — matrix operand order repaired and checked by reading, not compiled, not run on hardware
//
#include <hip/hip_runtime.h>

// ---------------------------------------------------------------------------
// CDNA5 (gfx1250, wave32) bf16 WMMA fragment helpers
// ---------------------------------------------------------------------------
typedef __attribute__((ext_vector_type(16))) __bf16 v16bf;
typedef __attribute__((ext_vector_type(8)))  __bf16 v8bf;
typedef __attribute__((ext_vector_type(8)))  float  v8f;

__device__ __forceinline__ v16bf mk16(v8bf lo, v8bf hi) {
  v16bf r;
#pragma unroll
  for (int e = 0; e < 8; ++e) { r[e] = lo[e]; r[8 + e] = hi[e]; }
  return r;
}

// A-matrix 16x32 bf16 fragment (ISA 7.12.2): lane holds row M=lane&15.
// lanes 0-15: K = k0+0..7 and k0+16..23 ; lanes 16-31: K = k0+8..15 and k0+24..31
__device__ __forceinline__ v16bf load_a_frag(const __bf16* base, int stride,
                                             int k0, int lane) {
  const __bf16* p = base + (size_t)(lane & 15) * stride + k0 + ((lane < 16) ? 0 : 8);
  v8bf lo = *(const v8bf*)p;
  v8bf hi = *(const v8bf*)(p + 16);
  return mk16(lo, hi);
}

// B-matrix 32x16 bf16 fragment: lane holds column N = n0+(lane&15);
// lanes 0-15 hold K = k0+0..7 and k0+16..23, lanes 16-31 hold K = k0+8..15 and k0+24..31.
// Weights stored N-major so each half is a contiguous 16-byte read.
__device__ __forceinline__ v16bf load_b_frag(const __bf16* base, int stride,
                                             int n0, int k0, int lane) {
  const __bf16* p = base + (size_t)(n0 + (lane & 15)) * stride + k0 + ((lane < 16) ? 0 : 8);
  v8bf lo = *(const v8bf*)p;
  v8bf hi = *(const v8bf*)(p + 16);
  return mk16(lo, hi);
}

__device__ __forceinline__ v8f wmma_bf16(v16bf a, v16bf b, v8f c) {
  // (neg_a, A, neg_b, B, c_mod, C, reuse_a, reuse_b)
  return __builtin_amdgcn_wmma_f32_16x16x32_bf16(false, a, false, b, (short)0, c,
                                                 false, false);
}

__device__ __forceinline__ v8f splat8(float v) {
  v8f r;
#pragma unroll
  for (int i = 0; i < 8; ++i) r[i] = v;
  return r;
}

__device__ __forceinline__ float sigmoidf_(float x) {
  return 1.0f / (1.0f + __expf(-x));
}

// ---------------------------------------------------------------------------
// Kernel 1: weight prep — fp32 -> bf16, transpose to N-major [N][K]
//   Wf/Uf/Wr/Ur : (1024 K x 4096 N) fp32  ->  (4096 x 1024) bf16
//   Wout        : (512 ch x 256 o)  fp32  ->  (256 x 512)  bf16
// ---------------------------------------------------------------------------
__global__ __launch_bounds__(256) void prep_kernel(
    const float* __restrict__ Wf, const float* __restrict__ Uf,
    const float* __restrict__ Wr, const float* __restrict__ Ur,
    const float* __restrict__ Wout,
    __bf16* __restrict__ WtF, __bf16* __restrict__ UtF,
    __bf16* __restrict__ WtR, __bf16* __restrict__ UtR,
    __bf16* __restrict__ WoutT) {
  const int NT = 4096 * 1024;
  for (int idx = blockIdx.x * blockDim.x + threadIdx.x; idx < NT;
       idx += gridDim.x * blockDim.x) {
    int k = idx >> 12, n = idx & 4095;      // src row-major (K x N), coalesced read
    int dst = n * 1024 + k;
    WtF[dst] = (__bf16)Wf[idx];
    UtF[dst] = (__bf16)Uf[idx];
    WtR[dst] = (__bf16)Wr[idx];
    UtR[dst] = (__bf16)Ur[idx];
  }
  const int NO = 512 * 256;
  for (int idx = blockIdx.x * blockDim.x + threadIdx.x; idx < NO;
       idx += gridDim.x * blockDim.x) {
    int ch = idx >> 8, o = idx & 255;
    WoutT[o * 512 + ch] = (__bf16)Wout[idx];
  }
}

// ---------------------------------------------------------------------------
// Kernel 2: fused bidirectional LSTM, entire 4-step recurrence on-chip.
// grid = (B/32 batch tiles, 2 directions), block = 256 threads (8 wave32).
//
// BM = 32: each wave owns 16 hidden columns x 4 gates and computes TWO 16-row
// M-tiles, so every B-fragment (weight) load from L2 feeds two WMMAs —
// doubling FLOP per L2 byte (weight streaming is the bandwidth ceiling).
//
// LDS = x_t tile (32x1024 bf16) + SINGLE h buffer (32x1024 bf16), rows padded
// +8 bf16 (~132 KB total).  Instead of double-buffering h, the new h values
// are stashed in registers during the compute phase and written to LDS only
// after a barrier proves every wave finished reading h_{t-1}.  x_{s+1}
// staging shares that same write phase.  c-state lives entirely in VGPRs
// (its natural layout IS the WMMA D-fragment layout).
// h2 is scattered to global in the permuted layout the output GEMM consumes.
// ---------------------------------------------------------------------------
#define XROW 1032  // 1024 + 8 pad (keeps 16B alignment, shifts banks per row)

__device__ __forceinline__ void stage_x(const float* __restrict__ x,
                                        __bf16* __restrict__ xs,
                                        int b0, int s, int dir, int tid) {
  // fwd_in[s,b,h*256+c] = x[b, h, w=s, (rev? 255-c : c)]
  for (int idx = tid; idx < 32 * 1024; idx += 256) {
    int row = idx >> 10, d = idx & 1023;
    int hh = d >> 8, cc = d & 255;
    int c2 = dir ? (255 - cc) : cc;
    float v = x[(((size_t)(b0 + row) * 4 + hh) * 4 + s) * 256 + c2];
    xs[row * XROW + d] = (__bf16)v;
  }
}

__global__ __launch_bounds__(256) void lstm_kernel(
    const float* __restrict__ x,
    const __bf16* __restrict__ WtF, const __bf16* __restrict__ UtF,
    const float* __restrict__ bf_,
    const __bf16* __restrict__ WtR, const __bf16* __restrict__ UtR,
    const float* __restrict__ br_,
    __bf16* __restrict__ Rbuf) {
  extern __shared__ __bf16 smem[];
  __bf16* xs = smem;              // 32 x XROW   (x_t tile)
  __bf16* hb = xs + 32 * XROW;    // 32 x XROW   (h, single buffer)

  const int tid = threadIdx.x, lane = tid & 31, wave = tid >> 5;
  const int b0  = blockIdx.x * 32;
  const int dir = blockIdx.y;

  const __bf16* Wt   = dir ? WtR : WtF;
  const __bf16* Ut   = dir ? UtR : UtF;
  const float*  bias = dir ? br_ : bf_;

  for (int i = tid; i < 32 * XROW; i += 256) hb[i] = (__bf16)0.0f;

  v8f  cst[16];                    // cell state: [chunk p][M-tile t]
  v8bf hst[16];                    // h stash for deferred LDS write
#pragma unroll
  for (int i = 0; i < 16; ++i) cst[i] = splat8(0.0f);

  stage_x(x, xs, b0, 0, dir, tid);
  __syncthreads();

  for (int s = 0; s < 4; ++s) {
    // ================= compute phase (reads xs, hb; no LDS writes) =========
#pragma unroll
    for (int p = 0; p < 8; ++p) {
      const int j0 = (p * 8 + wave) * 16;   // this wave's 16 hidden columns
      v8f acc[4][2];
#pragma unroll
      for (int g = 0; g < 4; ++g) {
        float bv = bias[g * 1024 + j0 + (lane & 15)];
        acc[g][0] = splat8(bv);
        acc[g][1] = splat8(bv);
      }

      // z += x_t @ W   (K = 1024); one B fragment feeds both M-tiles
      for (int k0 = 0; k0 < 1024; k0 += 32) {
        v16bf a0 = load_a_frag(xs, XROW, k0, lane);
        v16bf a1 = load_a_frag(xs + 16 * XROW, XROW, k0, lane);
        __builtin_prefetch(Wt + (size_t)(j0 + (lane & 15)) * 1024 + k0 + 128, 0, 1);
#pragma unroll
        for (int g = 0; g < 4; ++g) {
          v16bf b = load_b_frag(Wt, 1024, g * 1024 + j0, k0, lane);
          acc[g][0] = wmma_bf16(a0, b, acc[g][0]);
          acc[g][1] = wmma_bf16(a1, b, acc[g][1]);
        }
      }
      // z += h_{t-1} @ U   (K = 1024)
      for (int k0 = 0; k0 < 1024; k0 += 32) {
        v16bf a0 = load_a_frag(hb, XROW, k0, lane);
        v16bf a1 = load_a_frag(hb + 16 * XROW, XROW, k0, lane);
#pragma unroll
        for (int g = 0; g < 4; ++g) {
          v16bf b = load_b_frag(Ut, 1024, g * 1024 + j0, k0, lane);
          acc[g][0] = wmma_bf16(a0, b, acc[g][0]);
          acc[g][1] = wmma_bf16(a1, b, acc[g][1]);
        }
      }

      // ---- gate nonlinearities (D-fragment: M = r + 8*(lane>=16), N = lane&15)
      const int col = j0 + (lane & 15);
      const int co = col >> 2, jj = col & 3;
#pragma unroll
      for (int t = 0; t < 2; ++t) {
        v8f  cp = cst[p * 2 + t];
        v8bf hv;
#pragma unroll
        for (int r = 0; r < 8; ++r) {
          int mrow = t * 16 + r + ((lane >> 4) << 3);
          float zi = acc[0][t][r], zf = acc[1][t][r];
          float zg = acc[2][t][r], zo = acc[3][t][r];
          float cn = sigmoidf_(zf) * cp[r] + sigmoidf_(zi) * tanhf(zg);
          float hn = sigmoidf_(zo) * tanhf(cn);
          cp[r] = cn;
          hv[r] = (__bf16)hn;
          // scatter into output-GEMM layout: Rbuf[((b*4+jj)*4+s)*512 + dir*256 + co]
          size_t grow = (((size_t)(b0 + mrow) * 4 + jj) * 4 + s);
          Rbuf[grow * 512 + dir * 256 + co] = (__bf16)hn;
        }
        cst[p * 2 + t] = cp;
        hst[p * 2 + t] = hv;
      }
    }
    __syncthreads();   // all reads of xs / hb for step s complete

    // ================= write phase: h_s -> LDS, stage x_{s+1} ==============
#pragma unroll
    for (int p = 0; p < 8; ++p) {
      const int col = (p * 8 + wave) * 16 + (lane & 15);
#pragma unroll
      for (int t = 0; t < 2; ++t) {
        v8bf hv = hst[p * 2 + t];
#pragma unroll
        for (int r = 0; r < 8; ++r) {
          int mrow = t * 16 + r + ((lane >> 4) << 3);
          hb[mrow * XROW + col] = hv[r];
        }
      }
    }
    if (s < 3) stage_x(x, xs, b0, s + 1, dir, tid);
    __syncthreads();   // h_s visible / x_{s+1} staged before next compute
  }
}

// ---------------------------------------------------------------------------
// Kernel 3: output 1x1 conv as bf16 WMMA GEMM (65536 x 512) @ (512 x 256)
// with fused bias + LeakyReLU(0.3) + gamma*y + x residual.
// grid = 4096 M-tiles, block = 256 threads (8 waves); wave w covers N-tiles
// {w, w+8}; K = 512 -> 16 WMMA per tile.
// ---------------------------------------------------------------------------
__global__ __launch_bounds__(256) void out_kernel(
    const __bf16* __restrict__ Rbuf, const __bf16* __restrict__ WoutT,
    const float* __restrict__ bout, const float* __restrict__ x,
    const float* __restrict__ gamma, float* __restrict__ out) {
  const int tid = threadIdx.x, lane = tid & 31, wave = tid >> 5;
  const int m0 = blockIdx.x * 16;
  const float gv = gamma[0];

#pragma unroll 1
  for (int t = 0; t < 2; ++t) {
    const int n0 = (wave + t * 8) * 16;
    v8f acc = splat8(bout[n0 + (lane & 15)]);
    for (int k0 = 0; k0 < 512; k0 += 32) {
      v16bf a = load_a_frag(Rbuf + (size_t)m0 * 512, 512, k0, lane);
      v16bf b = load_b_frag(WoutT, 512, n0, k0, lane);
      acc = wmma_bf16(a, b, acc);
    }
#pragma unroll
    for (int r = 0; r < 8; ++r) {
      int m = m0 + r + ((lane >> 4) << 3);
      int n = n0 + (lane & 15);
      float y = acc[r];
      y = y > 0.0f ? y : 0.3f * y;              // LeakyReLU alpha=0.3
      size_t off = (size_t)m * 256 + n;         // m == ((b*4+j)*4+s), NHWC x
      out[off] = gv * y + x[off];
    }
  }
}

// ---------------------------------------------------------------------------
// launch
// ---------------------------------------------------------------------------
extern "C" void kernel_launch(void* const* d_in, const int* in_sizes, int n_in,
                              void* d_out, int out_size, void* d_ws, size_t ws_size,
                              hipStream_t stream) {
  (void)in_sizes; (void)n_in; (void)out_size; (void)ws_size;

  const float* x     = (const float*)d_in[0];
  const float* Wf    = (const float*)d_in[1];
  const float* Uf    = (const float*)d_in[2];
  const float* bf_   = (const float*)d_in[3];
  const float* Wr    = (const float*)d_in[4];
  const float* Ur    = (const float*)d_in[5];
  const float* br_   = (const float*)d_in[6];
  const float* Wout  = (const float*)d_in[7];
  const float* bout  = (const float*)d_in[8];
  const float* gamma = (const float*)d_in[9];
  float* out = (float*)d_out;

  // workspace carve-up (~96.3 MB total)
  char* w = (char*)d_ws;
  size_t off = 0;
  __bf16* WtF = (__bf16*)(w + off); off += (size_t)4096 * 1024 * 2;
  __bf16* UtF = (__bf16*)(w + off); off += (size_t)4096 * 1024 * 2;
  __bf16* WtR = (__bf16*)(w + off); off += (size_t)4096 * 1024 * 2;
  __bf16* UtR = (__bf16*)(w + off); off += (size_t)4096 * 1024 * 2;
  __bf16* WoutT = (__bf16*)(w + off); off += (size_t)256 * 512 * 2;
  __bf16* Rbuf  = (__bf16*)(w + off); off += (size_t)65536 * 512 * 2;

  prep_kernel<<<dim3(2048), dim3(256), 0, stream>>>(Wf, Uf, Wr, Ur, Wout,
                                                    WtF, UtF, WtR, UtR, WoutT);

  size_t shmem = (size_t)2 * 32 * XROW * sizeof(__bf16);  // ~132 KB
  lstm_kernel<<<dim3(128, 2), dim3(256), shmem, stream>>>(
      x, WtF, UtF, bf_, WtR, UtR, br_, Rbuf);

  out_kernel<<<dim3(4096), dim3(256), 0, stream>>>(Rbuf, WoutT, bout, x, gamma, out);
}
